// SPD_NET_21698174780043
// MI455X (gfx1250) — matrix operand order repaired and checked by reading, not compiled, not run on hardware
//
#include <hip/hip_runtime.h>
#include <hip/hip_bf16.h>
#include <stdint.h>
#include <stddef.h>

// ---------------------------------------------------------------------------
// SPD-Net (21 SPDTransform layers) for MI455X / gfx1250 (wave32, WMMA).
// GEMMs (time-embedding MLPs) run as bf16 v_wmma_f32_16x16x32_bf16 with f32
// accumulate; per-sample SPD algebra + Jacobi eigendecomposition run as
// one-sample-per-lane fp32 VALU code.
// ---------------------------------------------------------------------------

#define BATCH 65536
#define TDIM  256
#define EPSV  1e-4f

typedef __bf16 bf16_t;
typedef __attribute__((ext_vector_type(16))) __bf16 v16bf;
typedef __attribute__((ext_vector_type(8)))  float  v8f;

struct __attribute__((aligned(16))) U4 { uint32_t a, b, c, d; };
struct __attribute__((aligned(16))) U8 { U4 lo, hi; };

static __device__ __forceinline__ bf16_t f2bf(float f) {
  uint32_t u = __builtin_bit_cast(uint32_t, f);
  u += 0x7FFFu + ((u >> 16) & 1u);            // round-to-nearest-even
  unsigned short s = (unsigned short)(u >> 16);
  return __builtin_bit_cast(bf16_t, s);
}

// 16-bit A-matrix 16x32 fragment: lanes 0-15 M=0..15 (K 0..7 | 16..23),
// lanes 16-31 M=0..15 (K 8..15 | 24..31).  Two 16B chunks per lane.
static __device__ __forceinline__ v16bf ld_a_frag(const bf16_t* row, int half) {
  U8 t;
  t.lo = *(const U4*)(row + half * 8);
  t.hi = *(const U4*)(row + 16 + half * 8);
  return __builtin_bit_cast(v16bf, t);
}

// B fragment pre-packed so each lane reads 16 contiguous bf16 (32B).
static __device__ __forceinline__ v16bf ld_b_frag(const bf16_t* p) {
  U8 t;
  t.lo = *(const U4*)(p);
  t.hi = *(const U4*)(p + 8);
  return __builtin_bit_cast(v16bf, t);
}

static __device__ __forceinline__ v8f wmma_bf16(v16bf a, v16bf b, v8f c) {
  return __builtin_amdgcn_wmma_f32_16x16x32_bf16(false, a, false, b,
                                                 (short)0, c, false, false);
}

// ---------------------------------------------------------------------------
// Positional encoding: temb[b, 0:128]=sin(t*invf), temb[b,128:256]=cos(...)
// ---------------------------------------------------------------------------
__global__ void __launch_bounds__(256) pos_enc(const float* __restrict__ t,
                                               bf16_t* __restrict__ temb) {
  int tid = blockIdx.x * 256 + threadIdx.x;       // BATCH*128 threads
  int b = tid >> 7, k = tid & 127;
  float tv = t[b];
  float freq = __expf(-9.210340371976184f * (float)(2 * k) * (1.0f / 256.0f));
  float ang = tv * freq;
  temb[(size_t)b * TDIM + k]        = f2bf(__sinf(ang));
  temb[(size_t)b * TDIM + 128 + k]  = f2bf(__cosf(ang));
}

// ---------------------------------------------------------------------------
// Repack fp32 weight W[K=256, N] into WMMA B-fragment layout (bf16).
// For B (KxN, 16-bit): lane = 16*? ; lanes 0-15 hold K=kt*32+{0..7,16..23} for col n,
// lanes 16-31 hold K=kt*32+{8..15,24..31}.  Packed: [((nt*8+kt)*32+lane)*16 + j].
// ---------------------------------------------------------------------------
__global__ void repack_w(const float* __restrict__ W, bf16_t* __restrict__ out,
                         int N, int NT) {
  int tid = blockIdx.x * 256 + threadIdx.x;
  if (tid >= NT * 8 * 32) return;
  int lane = tid & 31;
  int kt   = (tid >> 5) & 7;
  int nt   = tid >> 8;
  int n    = nt * 16 + (lane & 15);
  int half = lane >> 4;
  bf16_t* dst = out + (size_t)((nt * 8 + kt) * 32 + lane) * 16;
  for (int j = 0; j < 16; ++j) {
    int k = kt * 32 + (j & 8) * 2 + half * 8 + (j & 7);
    float v = (n < N) ? W[(size_t)k * N + n] : 0.0f;
    dst[j] = f2bf(v);
  }
}

// ---------------------------------------------------------------------------
// GEMM1: h = silu(temb @ w1 + b1), [B,256]x[256,256] bf16 WMMA -> bf16 h
// Each wave: 16(M) x 32(N) tile, 8 K-steps.
// ---------------------------------------------------------------------------
__global__ void __launch_bounds__(256) gemm1_silu(
    const bf16_t* __restrict__ temb, const bf16_t* __restrict__ w1p,
    const float* __restrict__ b1, bf16_t* __restrict__ h) {
  int gtid = blockIdx.x * 256 + threadIdx.x;
  int wave = gtid >> 5;
  int lane = threadIdx.x & 31;
  int mt  = wave >> 3;             // 0..4095 batch tile
  int ntp = wave & 7;              // N tile pair
  int n = lane & 15, half = lane >> 4;

  v8f acc0 = {}; v8f acc1 = {};
  const bf16_t* arow = temb + (size_t)(mt * 16 + n) * TDIM;
  const bf16_t* bp0  = w1p + (size_t)((2 * ntp + 0) * 8) * 32 * 16 + lane * 16;
  const bf16_t* bp1  = w1p + (size_t)((2 * ntp + 1) * 8) * 32 * 16 + lane * 16;
  __builtin_prefetch(arow, 0, 1);
#pragma unroll
  for (int kt = 0; kt < 8; ++kt) {
    v16bf a  = ld_a_frag(arow + kt * 32, half);
    v16bf b0 = ld_b_frag(bp0 + kt * 32 * 16);
    v16bf b1v = ld_b_frag(bp1 + kt * 32 * 16);
    acc0 = wmma_bf16(a, b0, acc0);
    acc1 = wmma_bf16(a, b1v, acc1);
  }
#pragma unroll
  for (int j = 0; j < 8; ++j) {
    int row = mt * 16 + half * 8 + j;       // C/D layout: VGPR j -> M=half*8+j
    int c0 = (2 * ntp + 0) * 16 + n;
    int c1 = (2 * ntp + 1) * 16 + n;
    float v0 = acc0[j] + b1[c0];
    float v1 = acc1[j] + b1[c1];
    v0 = v0 / (1.0f + __expf(-v0));
    v1 = v1 / (1.0f + __expf(-v1));
    h[(size_t)row * TDIM + c0] = f2bf(v0);
    h[(size_t)row * TDIM + c1] = f2bf(v1);
  }
}

// ---------------------------------------------------------------------------
// GEMM2: emb = h @ w2 + b2, [B,256]x[256,out2] -> fp32 emb (stride out2)
// ---------------------------------------------------------------------------
__global__ void __launch_bounds__(256) gemm2_emb(
    const bf16_t* __restrict__ h, const bf16_t* __restrict__ w2p,
    const float* __restrict__ b2, float* __restrict__ emb,
    int out2, int NT) {
  int gtid = blockIdx.x * 256 + threadIdx.x;
  int wave = gtid >> 5;
  int lane = threadIdx.x & 31;
  int mt = wave / NT;
  int nt = wave - mt * NT;
  int n = lane & 15, half = lane >> 4;

  v8f acc = {};
  const bf16_t* arow = h + (size_t)(mt * 16 + n) * TDIM;
  const bf16_t* bp   = w2p + (size_t)(nt * 8) * 32 * 16 + lane * 16;
  __builtin_prefetch(arow, 0, 1);
#pragma unroll
  for (int kt = 0; kt < 8; ++kt) {
    v16bf a = ld_a_frag(arow + kt * 32, half);
    v16bf b = ld_b_frag(bp + kt * 32 * 16);
    acc = wmma_bf16(a, b, acc);
  }
  int col = nt * 16 + n;
  if (col < out2) {
    float bias = b2[col];
#pragma unroll
    for (int j = 0; j < 8; ++j) {
      int row = mt * 16 + half * 8 + j;
      emb[(size_t)row * out2 + col] = acc[j] + bias;
    }
  }
}

// ---------------------------------------------------------------------------
// Init state (stride 64) from x [B,4,4]
// ---------------------------------------------------------------------------
__global__ void __launch_bounds__(256) init_state(const float* __restrict__ x,
                                                  float* __restrict__ state) {
  int tid = blockIdx.x * 256 + threadIdx.x;       // BATCH*16
  state[((size_t)(tid >> 4)) * 64 + (tid & 15)] = x[tid];
}

// ---------------------------------------------------------------------------
// Per-thread symmetric Jacobi eigendecomp -> V * max(|lam|, eps) * V^T
// ---------------------------------------------------------------------------
template <int N>
static __device__ void rectify_inplace(float* A) {
  float V[N * N];
#pragma unroll
  for (int i = 0; i < N; ++i)
#pragma unroll
    for (int j = 0; j < N; ++j) V[i * N + j] = (i == j) ? 1.0f : 0.0f;

  const int sweeps = (N == 2) ? 2 : ((N == 4) ? 6 : 9);
  for (int s = 0; s < sweeps; ++s) {
#pragma unroll
    for (int p = 0; p < N - 1; ++p) {
#pragma unroll
      for (int q = p + 1; q < N; ++q) {
        float apq = A[p * N + q];
        if (apq * apq > 1e-24f) {
          float app = A[p * N + p], aqq = A[q * N + q];
          float tau = (aqq - app) / (2.0f * apq);
          float sq = sqrtf(1.0f + tau * tau);
          float tt = (tau >= 0.0f) ? 1.0f / (tau + sq) : 1.0f / (tau - sq);
          float c  = rsqrtf(1.0f + tt * tt);
          float ss = tt * c;
#pragma unroll
          for (int k = 0; k < N; ++k) {
            float akp = A[k * N + p], akq = A[k * N + q];
            A[k * N + p] = c * akp - ss * akq;
            A[k * N + q] = ss * akp + c * akq;
          }
#pragma unroll
          for (int k = 0; k < N; ++k) {
            float apk = A[p * N + k], aqk = A[q * N + k];
            A[p * N + k] = c * apk - ss * aqk;
            A[q * N + k] = ss * apk + c * aqk;
          }
#pragma unroll
          for (int k = 0; k < N; ++k) {
            float vkp = V[k * N + p], vkq = V[k * N + q];
            V[k * N + p] = c * vkp - ss * vkq;
            V[k * N + q] = ss * vkp + c * vkq;
          }
        }
      }
    }
  }
  float sv[N];
#pragma unroll
  for (int i = 0; i < N; ++i) sv[i] = fmaxf(fabsf(A[i * N + i]), EPSV);
#pragma unroll
  for (int i = 0; i < N; ++i)
#pragma unroll
    for (int j = 0; j < N; ++j) {
      float acc = 0.0f;
#pragma unroll
      for (int k = 0; k < N; ++k) acc += V[i * N + k] * sv[k] * V[j * N + k];
      A[i * N + j] = acc;
    }
}

// ---------------------------------------------------------------------------
// Per-sample SPD transform: optional skip-mix, dim-increase, W^T x W,
// T y T^T, optional rectify, write state / skip / out.
// ---------------------------------------------------------------------------
template <int IN, int OUT, bool RECT>
__global__ void __launch_bounds__(256) spd_layer(
    const float* __restrict__ state_in, float* __restrict__ state_out,
    const float* __restrict__ emb, const float* __restrict__ W,
    const float* __restrict__ mix, float* __restrict__ save) {
  constexpr int E = (OUT > IN) ? OUT : IN;
  int b = blockIdx.x * 256 + threadIdx.x;

  float xin[IN * IN];
  const float* sp = state_in + (size_t)b * 64;
#pragma unroll
  for (int i = 0; i < IN * IN; ++i) xin[i] = sp[i];
  if (mix) {
    const float* mp = mix + (size_t)b * (IN * IN);
#pragma unroll
    for (int i = 0; i < IN * IN; ++i) xin[i] = 0.25f * (xin[i] + mp[i]);
  }

  float x[E * E];
#pragma unroll
  for (int i = 0; i < E; ++i)
#pragma unroll
    for (int j = 0; j < E; ++j)
      x[i * E + j] = (i < IN && j < IN) ? xin[i * IN + j]
                                        : ((i == j) ? 1.0f : 0.0f);

  float Wl[E * OUT];               // uniform -> scalar loads
#pragma unroll
  for (int i = 0; i < E * OUT; ++i) Wl[i] = W[i];

  float tmp[E * OUT];              // x @ W
#pragma unroll
  for (int i = 0; i < E; ++i)
#pragma unroll
    for (int o = 0; o < OUT; ++o) {
      float acc = 0.0f;
#pragma unroll
      for (int j = 0; j < E; ++j) acc += x[i * E + j] * Wl[j * OUT + o];
      tmp[i * OUT + o] = acc;
    }

  float y[OUT * OUT];              // W^T @ tmp
#pragma unroll
  for (int o = 0; o < OUT; ++o)
#pragma unroll
    for (int p = 0; p < OUT; ++p) {
      float acc = 0.0f;
#pragma unroll
      for (int i = 0; i < E; ++i) acc += Wl[i * OUT + o] * tmp[i * OUT + p];
      y[o * OUT + p] = acc;
    }

  float T[OUT * OUT];
  const float* ep = emb + (size_t)b * (OUT * OUT);
#pragma unroll
  for (int i = 0; i < OUT * OUT; ++i) T[i] = ep[i];

  float u[OUT * OUT];              // T @ y
#pragma unroll
  for (int i = 0; i < OUT; ++i)
#pragma unroll
    for (int k = 0; k < OUT; ++k) {
      float acc = 0.0f;
#pragma unroll
      for (int j = 0; j < OUT; ++j) acc += T[i * OUT + j] * y[j * OUT + k];
      u[i * OUT + k] = acc;
    }

  float z[OUT * OUT];              // u @ T^T
#pragma unroll
  for (int i = 0; i < OUT; ++i)
#pragma unroll
    for (int l = 0; l < OUT; ++l) {
      float acc = 0.0f;
#pragma unroll
      for (int k = 0; k < OUT; ++k) acc += u[i * OUT + k] * T[l * OUT + k];
      z[i * OUT + l] = acc;
    }

  if (RECT) rectify_inplace<OUT>(z);

  float* dp = state_out + (size_t)b * 64;
#pragma unroll
  for (int i = 0; i < OUT * OUT; ++i) dp[i] = z[i];
  if (save) {
    float* svp = save + (size_t)b * (OUT * OUT);
#pragma unroll
    for (int i = 0; i < OUT * OUT; ++i) svp[i] = z[i];
  }
}

// ---------------------------------------------------------------------------
// Host orchestration
// ---------------------------------------------------------------------------
struct LayerCfg { int in, out, mixIdx, saveIdx, rect; };
static const LayerCfg g_cfg[21] = {
  {4,8,-1,-1,1},{8,8,-1,-1,1},{8,8,-1,-1,1},{8,8,-1, 0,1},
  {8,4,-1,-1,1},{4,4,-1, 1,1},{4,4,-1,-1,1},{4,4,-1, 2,1},
  {4,2,-1,-1,1},{2,2,-1, 3,1},{2,2,-1,-1,1},{2,2,-1,-1,1},
  {2,4, 3,-1,1},{4,4,-1,-1,1},{4,4, 2,-1,1},{4,4,-1,-1,1},
  {4,8, 1,-1,1},{8,8,-1,-1,1},{8,8, 0,-1,1},{8,8,-1,-1,1},
  {8,8,-1,-1,0}
};

static void launch_spd(int i, const float* sin_, float* sout, const float* emb,
                       const float* W, const float* mix, float* save,
                       hipStream_t stream) {
  dim3 g(BATCH / 256), blk(256);
  switch (i) {
    case 0: case 16:
      spd_layer<4, 8, true ><<<g, blk, 0, stream>>>(sin_, sout, emb, W, mix, save); break;
    case 1: case 2: case 3: case 17: case 18: case 19:
      spd_layer<8, 8, true ><<<g, blk, 0, stream>>>(sin_, sout, emb, W, mix, save); break;
    case 4:
      spd_layer<8, 4, true ><<<g, blk, 0, stream>>>(sin_, sout, emb, W, mix, save); break;
    case 5: case 6: case 7: case 13: case 14: case 15:
      spd_layer<4, 4, true ><<<g, blk, 0, stream>>>(sin_, sout, emb, W, mix, save); break;
    case 8:
      spd_layer<4, 2, true ><<<g, blk, 0, stream>>>(sin_, sout, emb, W, mix, save); break;
    case 9: case 10: case 11:
      spd_layer<2, 2, true ><<<g, blk, 0, stream>>>(sin_, sout, emb, W, mix, save); break;
    case 12:
      spd_layer<2, 4, true ><<<g, blk, 0, stream>>>(sin_, sout, emb, W, mix, save); break;
    default: // 20
      spd_layer<8, 8, false><<<g, blk, 0, stream>>>(sin_, sout, emb, W, mix, save); break;
  }
}

extern "C" void kernel_launch(void* const* d_in, const int* in_sizes, int n_in,
                              void* d_out, int out_size, void* d_ws, size_t ws_size,
                              hipStream_t stream) {
  (void)in_sizes; (void)n_in; (void)out_size; (void)ws_size;
  const float* x = (const float*)d_in[0];
  const float* t = (const float*)d_in[1];

  // Workspace carve (aligned to 256B)
  size_t off = 0;
  auto alloc = [&](size_t bytes) -> void* {
    void* p = (void*)((char*)d_ws + off);
    off = (off + bytes + 255) & ~(size_t)255;
    return p;
  };
  bf16_t* temb = (bf16_t*)alloc((size_t)BATCH * TDIM * 2);
  bf16_t* hbuf = (bf16_t*)alloc((size_t)BATCH * TDIM * 2);
  float*  embb = (float*) alloc((size_t)BATCH * 64 * 4);
  float*  state= (float*) alloc((size_t)BATCH * 64 * 4);
  float*  skip[4];
  skip[0] = (float*)alloc((size_t)BATCH * 64 * 4);
  skip[1] = (float*)alloc((size_t)BATCH * 16 * 4);
  skip[2] = (float*)alloc((size_t)BATCH * 16 * 4);
  skip[3] = (float*)alloc((size_t)BATCH *  4 * 4);
  bf16_t* w1p[21]; bf16_t* w2p[21];
  for (int i = 0; i < 21; ++i) {
    w1p[i] = (bf16_t*)alloc((size_t)16 * 8 * 32 * 16 * 2);  // 128KB
    w2p[i] = (bf16_t*)alloc((size_t) 4 * 8 * 32 * 16 * 2);  //  32KB
  }

  pos_enc   <<<BATCH * 128 / 256, 256, 0, stream>>>(t, temb);
  init_state<<<BATCH *  16 / 256, 256, 0, stream>>>(x, state);

  // One-time weight repacks into WMMA B-fragment layout
  for (int i = 0; i < 21; ++i) {
    const float* w1 = (const float*)d_in[2 + 5 * i + 1];
    const float* w2 = (const float*)d_in[2 + 5 * i + 3];
    int out2 = g_cfg[i].out * g_cfg[i].out;
    int NT2 = (out2 + 15) / 16;
    repack_w<<<(16 * 8 * 32 + 255) / 256, 256, 0, stream>>>(w1, w1p[i], 256, 16);
    repack_w<<<(NT2 * 8 * 32 + 255) / 256, 256, 0, stream>>>(w2, w2p[i], out2, NT2);
  }

  // 21 layers
  for (int i = 0; i < 21; ++i) {
    const float* Wt = (const float*)d_in[2 + 5 * i + 0];
    const float* b1 = (const float*)d_in[2 + 5 * i + 2];
    const float* b2 = (const float*)d_in[2 + 5 * i + 4];
    int out2 = g_cfg[i].out * g_cfg[i].out;
    int NT2 = (out2 + 15) / 16;

    gemm1_silu<<<4096, 256, 0, stream>>>(temb, w1p[i], b1, hbuf);
    gemm2_emb <<<512 * NT2, 256, 0, stream>>>(hbuf, w2p[i], b2, embb, out2, NT2);

    const float* mixp = (g_cfg[i].mixIdx >= 0) ? skip[g_cfg[i].mixIdx] : nullptr;
    float* savep = (g_cfg[i].saveIdx >= 0) ? skip[g_cfg[i].saveIdx] : nullptr;
    float* dst = (i == 20) ? (float*)d_out : state;
    launch_spd(i, state, dst, embb, Wt, mixp, savep, stream);
  }
}
